// MultiViewCT_51719996178473
// MI455X (gfx1250) — compile-verified, matrix operand order checked by reading, not run on hardware
//
#include <hip/hip_runtime.h>
#include <hip/hip_bf16.h>

typedef _Float16 v16h __attribute__((ext_vector_type(16)));
typedef _Float16 v8h  __attribute__((ext_vector_type(8)));
typedef float    v8f  __attribute__((ext_vector_type(8)));
typedef unsigned int u32x4 __attribute__((ext_vector_type(4)));
typedef int          i32x4 __attribute__((ext_vector_type(4)));
typedef int          i32x8 __attribute__((ext_vector_type(8)));

#define ZDIM 128
#define HDIM 256
#define WDIM 256
#define FEAT 128
#define NPTS_C 65536
#define BVN 8                    // B*V
#define BROWS (BVN * NPTS_C)     // 524288 view-MLP rows
#define GROWS (2 * NPTS_C)       // 131072 global-MLP rows
#define KIN 160                  // 137 padded to multiple of 32

// ---- workspace byte offsets --------------------------------------------
#define OFF_WINT   ((size_t)0)                                   // 128x160 f16
#define OFF_W1T(i) ((size_t)40960  + (size_t)(i) * 32768)        // 3 x 128x128
#define OFF_W2T(i) ((size_t)139264 + (size_t)(i) * 32768)        // 3 x 128x128
#define OFF_WOUTT  ((size_t)237568)                              // 128x128
#define OFF_GINT   ((size_t)270336)                              // 128x128
#define OFF_G1T(i) ((size_t)303104 + (size_t)(i) * 32768)        // 2 x 128x128
#define OFF_G2T(i) ((size_t)368640 + (size_t)(i) * 32768)        // 2 x 128x128
#define OFF_FEATS  ((size_t)1 << 20)
#define OFF_XIN    (OFF_FEATS + (size_t)BVN * HDIM * WDIM * FEAT * 2)
#define OFF_R      (OFF_XIN   + (size_t)BROWS * KIN * 2)
#define OFF_XG     (OFF_R     + (size_t)BROWS * FEAT * 2)

// ---- weight transpose/pad to f16 (dst[n*Kpad+k] = src[k*N+n]) ----------
__global__ void transpose_pad_f16(const float* __restrict__ src,
                                  _Float16* __restrict__ dst,
                                  int K, int N, int Kpad) {
    int id = blockIdx.x * 256 + threadIdx.x;
    if (id >= N * Kpad) return;
    int n = id / Kpad, k = id % Kpad;
    dst[id] = (k < K) ? (_Float16)src[(size_t)k * N + n] : (_Float16)0.f;
}

// ---- 3x3 conv, 1->128 ch, ReLU, f16 channel-last output ----------------
__global__ __launch_bounds__(256)
void conv_kernel(const float* __restrict__ views, const float* __restrict__ cw,
                 const float* __restrict__ cb, _Float16* __restrict__ feats) {
    __shared__ float sIn[12];                    // 3 rows x 4 cols patch
    int tid = threadIdx.x;
    int id  = blockIdx.x;                        // 8 imgs * 32768 pixel-pairs
    int bv  = id >> 15;
    int rem = id & 32767;
    int y   = rem >> 7;
    int x   = (rem & 127) << 1;
    const float* img = views + (size_t)bv * HDIM * WDIM;
    if (tid < 12) {
        int r = tid >> 2, cc = tid & 3;
        int yy = y - 1 + r, xx = x - 1 + cc;
        float v = 0.f;
        if (yy >= 0 && yy < HDIM && xx >= 0 && xx < WDIM) v = img[yy * WDIM + xx];
        sIn[tid] = v;
    }
    __syncthreads();
    int c = tid & 127, p = tid >> 7;
    float acc = cb[c];
#pragma unroll
    for (int ky = 0; ky < 3; ++ky)
#pragma unroll
        for (int kx = 0; kx < 3; ++kx)
            acc = fmaf(cw[c * 9 + ky * 3 + kx], sIn[ky * 4 + kx + p], acc);
    acc = fmaxf(acc, 0.f);
    feats[(((size_t)bv * HDIM + y) * WDIM + (x + p)) * FEAT + c] = (_Float16)acc;
}

// ---- projection + bilinear sample + PE -> xin rows [BROWS][160] f16 ----
__global__ __launch_bounds__(256)
void sample_kernel(const _Float16* __restrict__ feats,
                   const float* __restrict__ angles,
                   const int* __restrict__ idx,
                   _Float16* __restrict__ xin) {
    int tid = threadIdx.x;
    int q   = blockIdx.x * 2 + (tid >> 7);       // global row, 2 points/block
    int c   = tid & 127;
    int bv  = q >> 16;
    int pt  = q & (NPTS_C - 1);
    int gi  = idx[pt];
    int iz = gi >> 14, iy = (gi >> 7) & 127, ix = gi & 127;
    const float inv = 2.f / 127.f;
    float z  = iz * inv - 1.f;
    float yc = iy * inv - 1.f;
    float xc = ix * inv - 1.f;
    float a  = angles[bv];
    float ct = __cosf(a), st = __sinf(a);
    float u  = xc * ct - yc * st;
    float px = (u + 1.f) * 0.5f * (float)(WDIM - 1);
    float py = (z + 1.f) * 0.5f * (float)(HDIM - 1);
    px = fminf(fmaxf(px, 0.f), (float)(WDIM - 1));
    py = fminf(fmaxf(py, 0.f), (float)(HDIM - 1));
    int x0 = min(max((int)floorf(px), 0), WDIM - 2);
    int y0 = min(max((int)floorf(py), 0), HDIM - 2);
    float wx = px - (float)x0, wy = py - (float)y0;
    const _Float16* base = feats + (((size_t)bv * HDIM + y0) * WDIM + x0) * FEAT;
    float f00 = (float)base[c];
    float f01 = (float)base[FEAT + c];
    float f10 = (float)base[(size_t)WDIM * FEAT + c];
    float f11 = (float)base[(size_t)WDIM * FEAT + FEAT + c];
    float top = f00 * (1.f - wx) + f01 * wx;
    float bot = f10 * (1.f - wx) + f11 * wx;
    float f   = top * (1.f - wy) + bot * wy;
    size_t row = (size_t)q * KIN;
    xin[row + c] = (_Float16)f;
    if (c < 32) {                                // PE cols 128..136, pad ->159
        const float PI = 3.14159265358979f;
        float pe = 0.f;
        if (c < 3)      pe = (c == 0) ? z : (c == 1) ? yc : xc;
        else if (c < 6) { float t = (c == 3) ? z : (c == 4) ? yc : xc; pe = __sinf(PI * t); }
        else if (c < 9) { float t = (c == 6) ? z : (c == 7) ? yc : xc; pe = __cosf(PI * t); }
        xin[row + 128 + c] = (_Float16)pe;
    }
}

// ---- Tensor Data Mover: DMA a contiguous f16 block global -> LDS -------
// D# per ISA ch.8: 1-row 2D tile, data_size=2B, type=2.  Tracked by TENSORcnt.
__device__ __forceinline__ void tdm_load(const void* gsrc, void* lds_dst,
                                         unsigned nelem /* f16 elements */) {
    unsigned long long ga = (unsigned long long)(uintptr_t)gsrc;
    unsigned lo = (unsigned)(uintptr_t)lds_dst;  // LDS byte address (low 32b)
    u32x4 g0 = { 1u,                                   // count=1
                 lo,                                   // lds_addr
                 (unsigned)ga,                         // global_addr[31:0]
                 (unsigned)(ga >> 32) | 0x80000000u }; // ga[56:32] | type=2
    i32x8 g1;
    g1[0] = (int)(1u << 16);                     // workgroup_mask=0, data_size=1 (2B)
    g1[1] = (int)((nelem & 0xffffu) << 16);      // tensor_dim0[15:0]
    g1[2] = (int)((nelem >> 16) | (1u << 16));   // tensor_dim0[31:16], tensor_dim1=1
    g1[3] = (int)((nelem & 0xffffu) << 16);      // tile_dim0 = nelem
    g1[4] = 1;                                   // tile_dim1=1, tile_dim2=0
    g1[5] = (int)nelem;                          // tensor_dim0_stride[31:0]
    g1[6] = 0;
    g1[7] = 0;
    i32x4 z4 = {0, 0, 0, 0};
#if __clang_major__ >= 23
    i32x8 z8 = {0, 0, 0, 0, 0, 0, 0, 0};
    __builtin_amdgcn_tensor_load_to_lds(g0, g1, z4, z4, z8, 0);
#else
    __builtin_amdgcn_tensor_load_to_lds(g0, g1, z4, z4, 0);
#endif
}

// ---- WMMA helpers -------------------------------------------------------
// 16-bit A/B fragment (ISA 7.12.2): lane m=lane&15, hh=lane>>4;
// halves 0..7  = row (base+m), K = k0 + hh*8 .. +7
// halves 8..15 = row (base+m), K = k0 + 16 + hh*8 .. +7
__device__ __forceinline__ v16h load_frag(const _Float16* base, int rowBase,
                                          int stride, int k0, int lane) {
    int m = lane & 15, hh = (lane >> 4) & 1;
    const _Float16* p0 = base + (rowBase + m) * stride + k0 + hh * 8;
    v8h lo = *(const v8h*)p0;
    v8h hi = *(const v8h*)(p0 + 16);
    return __builtin_shufflevector(lo, hi, 0, 1, 2, 3, 4, 5, 6, 7,
                                   8, 9, 10, 11, 12, 13, 14, 15);
}

__device__ __forceinline__ void copy_lds(const void* g, void* l, int bytes,
                                         int tid, int nthr) {
    const uint4* s = (const uint4*)g;
    uint4* d = (uint4*)l;
    int n = bytes >> 4;
    for (int i = tid; i < n; i += nthr) d[i] = s[i];
}

// One layer tile: each wave owns 32 rows (2 M-tiles) so every B fragment is
// reused for 2 WMMAs (1.25 ds_load per wmma).  D = relu?(A*Wt^T + b [+resid])
template <int KTOT, bool RELU, bool RESID, bool GOUT>
__device__ __forceinline__ void gemm_layer(
    const _Float16* __restrict__ A, int sA,
    const _Float16* __restrict__ Wt, int sW,
    const float* __restrict__ bias,
    _Float16* D, int sD,
    const _Float16* Rsd, int sR,
    _Float16* gout, int rowBaseG,
    int lane, int waveRow) {
    v8f acc[2][8] = {};
#pragma unroll
    for (int k0 = 0; k0 < KTOT; k0 += 32) {
        v16h a0 = load_frag(A, waveRow,      sA, k0, lane);
        v16h a1 = load_frag(A, waveRow + 16, sA, k0, lane);
#pragma unroll
        for (int nt = 0; nt < 8; ++nt) {
            v16h b = load_frag(Wt, nt * 16, sW, k0, lane);
            acc[0][nt] = __builtin_amdgcn_wmma_f32_16x16x32_f16(
                false, a0, false, b, (short)0, acc[0][nt], false, false);
            acc[1][nt] = __builtin_amdgcn_wmma_f32_16x16x32_f16(
                false, a1, false, b, (short)0, acc[1][nt], false, false);
        }
    }
    int nl = lane & 15, hh = lane >> 4;
#pragma unroll
    for (int mt = 0; mt < 2; ++mt)
#pragma unroll
        for (int nt = 0; nt < 8; ++nt) {
            int col = nt * 16 + nl;
            float bb = bias[col];
#pragma unroll
            for (int r = 0; r < 8; ++r) {
                int row = waveRow + mt * 16 + r + 8 * hh;
                float v = acc[mt][nt][r] + bb;
                if (RESID) v += (float)Rsd[row * sR + col];
                if (RELU)  v = fmaxf(v, 0.f);
                if (GOUT)  gout[(size_t)(rowBaseG + row) * 128 + col] = (_Float16)v;
                else       D[row * sD + col] = (_Float16)v;
            }
        }
}

// ---- fused view MLP: 137->128, 3 residual blocks, 128->128 -------------
// 128 threads = 4 waves x 32 rows.  TDM double-buffers next layer's weights.
__global__ __launch_bounds__(128)
void view_mlp_kernel(const _Float16* __restrict__ xin, const char* __restrict__ wsb,
                     const float* __restrict__ vb_in, const float* __restrict__ vb1,
                     const float* __restrict__ vb2, const float* __restrict__ vb_out,
                     _Float16* __restrict__ rout) {
    __shared__ __align__(16) _Float16 sA[128 * KIN];   // input, then T buffer
    __shared__ __align__(16) _Float16 sH[128 * 128];   // hidden state
    __shared__ __align__(16) _Float16 sW0[128 * KIN];  // weight ping (Win, W2_*)
    __shared__ __align__(16) _Float16 sW1[128 * 128];  // weight pong (W1_*, Wout)
    int tid = threadIdx.x, lane = tid & 31, wr = (tid >> 5) * 32;
    bool w0 = (tid < 32);
    int rowBase = blockIdx.x * 128;

    if (w0) tdm_load(wsb + OFF_WINT, sW0, 128 * KIN);          // Win DMA
    copy_lds(xin + (size_t)rowBase * KIN, sA, 128 * KIN * 2, tid, 128);
    if (w0) __builtin_amdgcn_s_wait_tensorcnt(0);
    __syncthreads();

    if (w0) tdm_load(wsb + OFF_W1T(0), sW1, 128 * 128);        // prefetch L1
    gemm_layer<KIN, true, false, false>(sA, KIN, sW0, KIN, vb_in,
                                        sH, 128, nullptr, 0, nullptr, 0, lane, wr);
    if (w0) __builtin_amdgcn_s_wait_tensorcnt(0);
    __syncthreads();

    const size_t offs[7] = {OFF_W1T(0), OFF_W2T(0), OFF_W1T(1), OFF_W2T(1),
                            OFF_W1T(2), OFF_W2T(2), OFF_WOUTT};
    for (int j = 0; j < 7; ++j) {
        const _Float16* cur = (j & 1) ? sW0 : sW1;
        _Float16*       nxt = (j & 1) ? sW1 : sW0;
        if (w0 && j < 6) tdm_load(wsb + offs[j + 1], nxt, 128 * 128);
        if (j == 6) {
            gemm_layer<128, false, false, true>(sH, 128, cur, 128, vb_out,
                nullptr, 0, nullptr, 0, rout, rowBase, lane, wr);
        } else if ((j & 1) == 0) {            // t = relu(h@W1 + b1) -> sA
            gemm_layer<128, true, false, false>(sH, 128, cur, 128, vb1 + (j >> 1) * 128,
                sA, KIN, nullptr, 0, nullptr, 0, lane, wr);
        } else {                              // h = relu(h + t@W2 + b2) -> sH
            gemm_layer<128, true, true, false>(sA, KIN, cur, 128, vb2 + (j >> 1) * 128,
                sH, 128, sH, 128, nullptr, 0, lane, wr);
        }
        if (w0) __builtin_amdgcn_s_wait_tensorcnt(0);
        __syncthreads();
    }
}

// ---- mean over 4 views: r[BV][NPTS][128] -> xg[B*NPTS][128] ------------
__global__ __launch_bounds__(256)
void mean_kernel(const _Float16* __restrict__ r, _Float16* __restrict__ xg) {
    int e = blockIdx.x * 256 + threadIdx.x;      // (b<<23)|(pt<<7)|c
    int c = e & 127, pt = (e >> 7) & (NPTS_C - 1), b = e >> 23;
    float s = 0.f;
#pragma unroll
    for (int v = 0; v < 4; ++v)
        s += (float)r[(size_t)((b * 4 + v) * NPTS_C + pt) * 128 + c];
    xg[(size_t)e] = (_Float16)(0.25f * s);
}

// ---- fused global MLP: 128->128, 2 residual blocks, 128->1 -------------
__global__ __launch_bounds__(128)
void global_mlp_kernel(const _Float16* __restrict__ xg, const char* __restrict__ wsb,
                       const float* __restrict__ gb_in, const float* __restrict__ gb1,
                       const float* __restrict__ gb2, const float* __restrict__ gw_out,
                       const float* __restrict__ gb_out, float* __restrict__ out) {
    __shared__ __align__(16) _Float16 sA[128 * 128];
    __shared__ __align__(16) _Float16 sH[128 * 128];
    __shared__ __align__(16) _Float16 sW0[128 * 128];
    __shared__ __align__(16) _Float16 sW1[128 * 128];
    int tid = threadIdx.x, lane = tid & 31, wr = (tid >> 5) * 32;
    bool w0 = (tid < 32);
    int rowBase = blockIdx.x * 128;

    if (w0) tdm_load(wsb + OFF_GINT, sW0, 128 * 128);
    copy_lds(xg + (size_t)rowBase * 128, sA, 128 * 128 * 2, tid, 128);
    if (w0) __builtin_amdgcn_s_wait_tensorcnt(0);
    __syncthreads();

    if (w0) tdm_load(wsb + OFF_G1T(0), sW1, 128 * 128);
    gemm_layer<128, true, false, false>(sA, 128, sW0, 128, gb_in,
                                        sH, 128, nullptr, 0, nullptr, 0, lane, wr);
    if (w0) __builtin_amdgcn_s_wait_tensorcnt(0);
    __syncthreads();

    const size_t offs[4] = {OFF_G1T(0), OFF_G2T(0), OFF_G1T(1), OFF_G2T(1)};
    for (int j = 0; j < 4; ++j) {
        const _Float16* cur = (j & 1) ? sW0 : sW1;
        _Float16*       nxt = (j & 1) ? sW1 : sW0;
        if (w0 && j < 3) tdm_load(wsb + offs[j + 1], nxt, 128 * 128);
        if ((j & 1) == 0) {
            gemm_layer<128, true, false, false>(sH, 128, cur, 128, gb1 + (j >> 1) * 128,
                sA, 128, nullptr, 0, nullptr, 0, lane, wr);
        } else {
            gemm_layer<128, true, true, false>(sA, 128, cur, 128, gb2 + (j >> 1) * 128,
                sH, 128, sH, 128, nullptr, 0, lane, wr);
        }
        if (w0) __builtin_amdgcn_s_wait_tensorcnt(0);
        __syncthreads();
    }
    if (tid < 128) {                             // final 128 -> 1 dot
        float s = gb_out[0];
        for (int k = 0; k < 128; ++k)
            s += (float)sH[tid * 128 + k] * gw_out[k];
        out[rowBase + tid] = s;
    }
}

// ---- host launcher ------------------------------------------------------
extern "C" void kernel_launch(void* const* d_in, const int* in_sizes, int n_in,
                              void* d_out, int out_size, void* d_ws, size_t ws_size,
                              hipStream_t stream) {
    const float* views  = (const float*)d_in[0];
    const float* angles = (const float*)d_in[1];
    const int*   idx    = (const int*)  d_in[2];
    const float* conv_w = (const float*)d_in[3];
    const float* conv_b = (const float*)d_in[4];
    const float* vw_in  = (const float*)d_in[5];
    const float* vb_in  = (const float*)d_in[6];
    const float* vw1    = (const float*)d_in[7];
    const float* vb1    = (const float*)d_in[8];
    const float* vw2    = (const float*)d_in[9];
    const float* vb2    = (const float*)d_in[10];
    const float* vw_out = (const float*)d_in[11];
    const float* vb_out = (const float*)d_in[12];
    const float* gw_in  = (const float*)d_in[13];
    const float* gb_in  = (const float*)d_in[14];
    const float* gw1    = (const float*)d_in[15];
    const float* gb1    = (const float*)d_in[16];
    const float* gw2    = (const float*)d_in[17];
    const float* gb2    = (const float*)d_in[18];
    const float* gw_out = (const float*)d_in[19];
    const float* gb_out = (const float*)d_in[20];
    char* ws = (char*)d_ws;

    _Float16* feats = (_Float16*)(ws + OFF_FEATS);
    _Float16* xin   = (_Float16*)(ws + OFF_XIN);
    _Float16* rbuf  = (_Float16*)(ws + OFF_R);
    _Float16* xg    = (_Float16*)(ws + OFF_XG);

    auto T = [&](const float* src, size_t off, int K, int N, int Kpad) {
        int total = N * Kpad;
        transpose_pad_f16<<<dim3((total + 255) / 256), dim3(256), 0, stream>>>(
            src, (_Float16*)(ws + off), K, N, Kpad);
    };
    T(vw_in, OFF_WINT, 137, 128, KIN);
    for (int i = 0; i < 3; ++i) {
        T(vw1 + (size_t)i * 128 * 128, OFF_W1T(i), 128, 128, 128);
        T(vw2 + (size_t)i * 128 * 128, OFF_W2T(i), 128, 128, 128);
    }
    T(vw_out, OFF_WOUTT, 128, 128, 128);
    T(gw_in, OFF_GINT, 128, 128, 128);
    for (int i = 0; i < 2; ++i) {
        T(gw1 + (size_t)i * 128 * 128, OFF_G1T(i), 128, 128, 128);
        T(gw2 + (size_t)i * 128 * 128, OFF_G2T(i), 128, 128, 128);
    }

    conv_kernel<<<dim3(BVN * HDIM * WDIM / 2), dim3(256), 0, stream>>>(
        views, conv_w, conv_b, feats);
    sample_kernel<<<dim3(BROWS / 2), dim3(256), 0, stream>>>(
        feats, angles, idx, xin);
    view_mlp_kernel<<<dim3(BROWS / 128), dim3(128), 0, stream>>>(
        xin, ws, vb_in, vb1, vb2, vb_out, rbuf);
    mean_kernel<<<dim3(GROWS * 128 / 256), dim3(256), 0, stream>>>(rbuf, xg);
    global_mlp_kernel<<<dim3(GROWS / 128), dim3(128), 0, stream>>>(
        xg, ws, gb_in, gb1, gb2, gw_out, gb_out, (float*)d_out);
}
